// ProcessorCNN_50663434224158
// MI455X (gfx1250) — matrix operand order repaired and checked by reading, not compiled, not run on hardware
//
#include <hip/hip_runtime.h>
#include <hip/hip_bf16.h>

// ---------------------------------------------------------------------------
// ProcessorCNN on MI455X (gfx1250): 3 steps of
//   x = relu( LN( x + x@Ws + mean6(neigh)@Wn + (bs+bn) ) * scale + off )
// GEMMs fused as [x | mean] (N x 512) @ Wt (512 x 256) using
// v_wmma_f32_16x16x32_bf16 (bf16 inputs, f32 accumulate).
// Weight panel staged to LDS via async-copy (ASYNCcnt) when available.
// ---------------------------------------------------------------------------

#define N_NODES 50000
#define N_PAD   50048          // 391 * 128
#define LATENT  256
#define KDIM    512            // 2*LATENT (x half + neighbor-mean half)
#define N_STEPS 3
#define BCOL_PAD 520           // 512 + 8 ushort pad -> 260 words, stride%64==4

typedef __attribute__((ext_vector_type(16))) __bf16 v16bf;
typedef __attribute__((ext_vector_type(8)))  float  v8f;
typedef __attribute__((ext_vector_type(4)))  unsigned int v4u;
typedef __attribute__((ext_vector_type(4)))  int v4i;

union Frag16 {
  v16bf bf;
  v4u   u[2];
};

#if defined(__has_builtin)
#if __has_builtin(__builtin_amdgcn_global_load_async_to_lds_b128)
#define HAVE_ASYNC_LDS 1
#endif
#if __has_builtin(__builtin_amdgcn_s_wait_asynccnt)
#define HAVE_WAIT_ASYNC 1
#endif
#endif

static __device__ __forceinline__ unsigned short f2bf(float f) {
  unsigned int u = __float_as_uint(f);
  unsigned int r = u + 0x7FFFu + ((u >> 16) & 1u);   // round-to-nearest-even
  return (unsigned short)(r >> 16);
}

static __device__ __forceinline__ unsigned long long pack4bf(float a, float b,
                                                             float c, float d) {
  return (unsigned long long)f2bf(a)
       | ((unsigned long long)f2bf(b) << 16)
       | ((unsigned long long)f2bf(c) << 32)
       | ((unsigned long long)f2bf(d) << 48);
}

// ---------------------------------------------------------------------------
// Prep: Wt[s][n][k] = (k<256 ? Ws[s][k][n] : Wn[s][k-256][n]) as bf16.
// ---------------------------------------------------------------------------
__global__ __launch_bounds__(256) void prep_wt(const float* __restrict__ Ws,
                                               const float* __restrict__ Wn,
                                               unsigned short* __restrict__ Wt) {
  int idx = blockIdx.x * 256 + threadIdx.x;      // 3 * 256 * 512 total
  int s   = idx >> 17;                           // / (256*512)
  int rem = idx & 131071;
  int n   = rem >> 9;                            // / 512
  int k   = rem & 511;
  float v = (k < LATENT) ? Ws[s * 65536 + k * LATENT + n]
                         : Wn[s * 65536 + (k - LATENT) * LATENT + n];
  Wt[idx] = f2bf(v);
}

__global__ __launch_bounds__(256) void prep_bias(const float* __restrict__ bs,
                                                 const float* __restrict__ bn,
                                                 float* __restrict__ bsum) {
  int idx = blockIdx.x * 256 + threadIdx.x;      // 3 * 256 total
  bsum[idx] = bs[idx] + bn[idx];
}

// ---------------------------------------------------------------------------
// Gather 6 neighbors, mean, and pack bf16 A-panel: A[r][0:256] = bf16(x[r]),
// A[r][256:512] = bf16(mean of 6 neighbor rows). 64 threads per row (float4).
// ---------------------------------------------------------------------------
__global__ __launch_bounds__(256) void gather_pack(
    const float* __restrict__ x, const int* __restrict__ nbr,
    unsigned short* __restrict__ A) {
  int gid = blockIdx.x * 256 + threadIdx.x;
  int r = gid >> 6;
  int c = gid & 63;                               // float4 index within row
  if (r >= N_NODES) return;

  const float4* xv = (const float4*)x;
  float4 self = xv[(size_t)r * 64 + c];

  int i0 = nbr[r * 6 + 0], i1 = nbr[r * 6 + 1], i2 = nbr[r * 6 + 2];
  int i3 = nbr[r * 6 + 3], i4 = nbr[r * 6 + 4], i5 = nbr[r * 6 + 5];
  float4 n0 = xv[(size_t)i0 * 64 + c];
  float4 n1 = xv[(size_t)i1 * 64 + c];
  float4 n2 = xv[(size_t)i2 * 64 + c];
  float4 n3 = xv[(size_t)i3 * 64 + c];
  float4 n4 = xv[(size_t)i4 * 64 + c];
  float4 n5 = xv[(size_t)i5 * 64 + c];
  const float inv6 = 1.0f / 6.0f;
  float mx = (n0.x + n1.x + n2.x + n3.x + n4.x + n5.x) * inv6;
  float my = (n0.y + n1.y + n2.y + n3.y + n4.y + n5.y) * inv6;
  float mz = (n0.z + n1.z + n2.z + n3.z + n4.z + n5.z) * inv6;
  float mw = (n0.w + n1.w + n2.w + n3.w + n4.w + n5.w) * inv6;

  unsigned short* arow = A + (size_t)r * KDIM;
  *(unsigned long long*)(arow + 4 * c) = pack4bf(self.x, self.y, self.z, self.w);
  *(unsigned long long*)(arow + LATENT + 4 * c) = pack4bf(mx, my, mz, mw);
}

// ---------------------------------------------------------------------------
// WMMA GEMM: G[r][n] = sum_k A[r][k] * Wt[n][k]   (bf16 x bf16 -> f32)
// Block = 256 threads (8 waves): 128 rows x 64 cols. Wave = 32x32 macro-tile.
// B panel (64 cols x 512 K bf16 = 64 KB) staged to LDS once per block via
// GLOBAL_LOAD_ASYNC_TO_LDS_B128; k-loop reads B with ds_load_b128.
// ---------------------------------------------------------------------------
__global__ __launch_bounds__(256) void wmma_gemm(
    const unsigned short* __restrict__ A,   // [N_PAD][512] bf16
    const unsigned short* __restrict__ Wt,  // [256][512]   bf16 (this step)
    float* __restrict__ G) {                // [N_NODES][256] f32
  __shared__ unsigned short Bs[64 * BCOL_PAD];   // 66,560 B

  const int tid   = threadIdx.x;
  const int wave  = tid >> 5;
  const int lane  = tid & 31;
  const int mwave = wave >> 1;              // 0..3 -> 32-row strip
  const int nwave = wave & 1;               // 0..1 -> 32-col strip
  const int lrow  = lane & 15;
  const int hi    = lane >> 4;              // 0: lanes 0-15, 1: lanes 16-31
  const int asel  = hi << 3;                // A: +8 K elements for upper lanes
  const int bsel  = hi << 3;                // B: +8 K elements for upper lanes

  const int r0 = blockIdx.x * 128 + mwave * 32;
  const int colBlock = blockIdx.y * 64;

  // ---- stage B panel: 64 cols x 512 K, 4096 chunks of 16 B ----
  {
    const unsigned short* Wb = Wt + (size_t)colBlock * KDIM;
#pragma unroll
    for (int i = 0; i < 16; ++i) {
      const int chunk = i * 256 + tid;
      const int col = chunk >> 6;           // 0..63
      const int seg = chunk & 63;           // 0..63 (8 ushorts each)
      const unsigned short* gp = Wb + col * KDIM + seg * 8;
      unsigned short* lp = &Bs[col * BCOL_PAD + seg * 8];
#if defined(HAVE_ASYNC_LDS)
      __builtin_amdgcn_global_load_async_to_lds_b128(
          (__attribute__((address_space(1))) v4i*)gp,
          (__attribute__((address_space(3))) v4i*)lp, 0, 0);
#else
      *(v4u*)lp = *(const v4u*)gp;
#endif
    }
#if defined(HAVE_ASYNC_LDS)
#if defined(HAVE_WAIT_ASYNC)
    __builtin_amdgcn_s_wait_asynccnt(0);
#else
    asm volatile("s_wait_asynccnt 0" ::: "memory");
#endif
#endif
    __syncthreads();
  }

  v8f acc[2][2] = {};

#pragma unroll 4
  for (int kt = 0; kt < 16; ++kt) {
    const int kb = kt * 32;
    Frag16 a[2], b[2];
#pragma unroll
    for (int t = 0; t < 2; ++t) {
      // A fragment (16x32 bf16): lane holds row (lrow), K = kb+asel..+7 and +16
      const unsigned short* ap = A + (size_t)(r0 + t * 16 + lrow) * KDIM + kb + asel;
      a[t].u[0] = *(const v4u*)ap;
      a[t].u[1] = *(const v4u*)(ap + 16);
      // B fragment (32x16 bf16) from LDS: lane holds col, K = kb+bsel..+7 and +16
      const unsigned short* bp =
          &Bs[(nwave * 32 + t * 16 + lrow) * BCOL_PAD + kb + bsel];
      b[t].u[0] = *(const v4u*)bp;
      b[t].u[1] = *(const v4u*)(bp + 16);
    }
#pragma unroll
    for (int mt = 0; mt < 2; ++mt)
#pragma unroll
      for (int nt = 0; nt < 2; ++nt)
        acc[mt][nt] = __builtin_amdgcn_wmma_f32_16x16x32_bf16(
            false, a[mt].bf, false, b[nt].bf, (short)0, acc[mt][nt],
            false, false);
  }

  // C/D layout: lanes 0-15: VGPR j -> M=j, N=lane; lanes 16-31: M=j+8, N=lane-16
#pragma unroll
  for (int mt = 0; mt < 2; ++mt) {
#pragma unroll
    for (int nt = 0; nt < 2; ++nt) {
      const int col = colBlock + nwave * 32 + nt * 16 + lrow;
#pragma unroll
      for (int j = 0; j < 8; ++j) {
        const int row = r0 + mt * 16 + (hi ? j + 8 : j);
        if (row < N_NODES) G[(size_t)row * LATENT + col] = acc[mt][nt][j];
      }
    }
  }
}

// ---------------------------------------------------------------------------
// Residual + bias + LayerNorm + ReLU.  One 256-thread block per row.
// Safe for xout == xin (each thread reads its element before writing it).
// ---------------------------------------------------------------------------
__global__ __launch_bounds__(256) void ln_relu(
    const float* __restrict__ xin, const float* __restrict__ G,
    const float* __restrict__ bsum, const float* __restrict__ scale,
    const float* __restrict__ offset, float* __restrict__ xout) {
  const int r = blockIdx.x;
  const int c = threadIdx.x;
  const size_t p = (size_t)r * LATENT + c;
  float v = xin[p] + G[p] + bsum[c];

  __shared__ float ssum[8];
  __shared__ float ssq[8];
  float s = v, q = v * v;
#pragma unroll
  for (int o = 16; o > 0; o >>= 1) {
    s += __shfl_xor(s, o, 32);
    q += __shfl_xor(q, o, 32);
  }
  if ((threadIdx.x & 31) == 0) {
    ssum[threadIdx.x >> 5] = s;
    ssq[threadIdx.x >> 5]  = q;
  }
  __syncthreads();
  float ts = 0.f, tq = 0.f;
#pragma unroll
  for (int i = 0; i < 8; ++i) { ts += ssum[i]; tq += ssq[i]; }

  const float mu  = ts * (1.0f / LATENT);
  const float var = tq * (1.0f / LATENT) - mu * mu;
  const float inv = rsqrtf(var + 1e-5f);
  float y = (v - mu) * inv * scale[c] + offset[c];
  xout[p] = fmaxf(y, 0.0f);
}

// ---------------------------------------------------------------------------
// Scratch layout (bytes):
//   A    : N_PAD * 512 * 2          = 51,249,152
//   Wt   : 3 * 256 * 512 * 2        =    786,432
//   bsum : 3 * 256 * 4              =      3,072
//   G    : N_NODES * 256 * 4        = 51,200,000
// total ~103.2 MB
// ---------------------------------------------------------------------------
extern "C" void kernel_launch(void* const* d_in, const int* in_sizes, int n_in,
                              void* d_out, int out_size, void* d_ws, size_t ws_size,
                              hipStream_t stream) {
  (void)in_sizes; (void)n_in; (void)out_size; (void)ws_size;

  const float* x0     = (const float*)d_in[0];   // sphere_nodes (N,256)
  const float* Ws     = (const float*)d_in[1];   // (3,256,256)
  const float* bs     = (const float*)d_in[2];   // (3,256)
  const float* Wn     = (const float*)d_in[3];   // (3,256,256)
  const float* bn     = (const float*)d_in[4];   // (3,256)
  const float* lsc    = (const float*)d_in[5];   // (3,256)
  const float* loff   = (const float*)d_in[6];   // (3,256)
  const int*   nbr    = (const int*)d_in[7];     // (N,6)

  char* ws = (char*)d_ws;
  const size_t A_OFF  = 0;
  const size_t A_SZ   = (size_t)N_PAD * KDIM * 2;
  const size_t WT_OFF = A_OFF + A_SZ;
  const size_t WT_SZ  = (size_t)N_STEPS * LATENT * KDIM * 2;
  const size_t BS_OFF = WT_OFF + WT_SZ;
  const size_t BS_SZ  = (size_t)N_STEPS * LATENT * 4;
  const size_t G_OFF  = BS_OFF + BS_SZ;

  unsigned short* A    = (unsigned short*)(ws + A_OFF);
  unsigned short* Wt   = (unsigned short*)(ws + WT_OFF);
  float*          bsum = (float*)(ws + BS_OFF);
  float*          G    = (float*)(ws + G_OFF);
  float*          out  = (float*)d_out;

  // Weight / bias prep (runs every call; deterministic).
  prep_wt<<<(N_STEPS * LATENT * KDIM) / 256, 256, 0, stream>>>(Ws, Wn, Wt);
  prep_bias<<<(N_STEPS * LATENT) / 256, 256, 0, stream>>>(bs, bn, bsum);

  const float* xcur = x0;
  for (int s = 0; s < N_STEPS; ++s) {
    gather_pack<<<(N_NODES * 64) / 256, 256, 0, stream>>>(xcur, nbr, A);
    dim3 ggrid(N_PAD / 128, LATENT / 64);
    wmma_gemm<<<ggrid, 256, 0, stream>>>(A, Wt + (size_t)s * LATENT * KDIM, G);
    ln_relu<<<N_NODES, 256, 0, stream>>>(xcur, G, bsum + s * LATENT,
                                         lsc + s * LATENT, loff + s * LATENT, out);
    xcur = out;  // steps 1,2 run in-place on d_out
  }
}
